// Inter_Patch_Attention_3985729651641
// MI455X (gfx1250) — matrix operand order repaired and checked by reading, not compiled, not run on hardware
//
#include <hip/hip_runtime.h>

// ---------------- problem constants ----------------
constexpr int Bb   = 4;
constexpr int Ll   = 2048;
constexpr int DM   = 512;   // d_model
constexpr int Hh   = 8;
constexpr int DK   = 64;
constexpr int Mm   = Bb * Ll;     // 8192 rows
constexpr int WIN  = 10;
constexpr float SCALE = 0.125f;   // 1/sqrt(64)
constexpr float CLAMP = 10000.0f;

// ---------------- vector types ----------------
typedef __attribute__((ext_vector_type(16))) __bf16 v16bf;
typedef __attribute__((ext_vector_type(8)))  __bf16 v8bf;
typedef __attribute__((ext_vector_type(8)))  float  v8f;
typedef __attribute__((ext_vector_type(4)))  unsigned int u32x4;

__device__ __forceinline__ v8bf load8(const __bf16* p) {
    return __builtin_bit_cast(v8bf, *(const u32x4*)p);
}
__device__ __forceinline__ void store8(__bf16* p, v8bf v) {
    *(u32x4*)p = __builtin_bit_cast(u32x4, v);
}
__device__ __forceinline__ v16bf frag(const __bf16* lo, const __bf16* hi) {
    v8bf a = load8(lo), b = load8(hi);
    return __builtin_shufflevector(a, b, 0,1,2,3,4,5,6,7,8,9,10,11,12,13,14,15);
}

// ---------------- prep kernels ----------------
__global__ void cvt_f32_bf16(const float* __restrict__ in, __bf16* __restrict__ out, int n) {
    int i = blockIdx.x * blockDim.x + threadIdx.x;
    if (i < n) out[i] = (__bf16)in[i];
}

// W stored [K=in][N=out]; produce Wt[N][K] in bf16 (coalesced read)
__global__ void transpose_cvt(const float* __restrict__ W, __bf16* __restrict__ Wt) {
    int flat = blockIdx.x * blockDim.x + threadIdx.x;   // 512*512 threads
    int k = flat >> 9;          // input dim
    int n = flat & 511;         // output dim
    Wt[n * DM + k] = (__bf16)W[k * DM + n];
}

// ---------------- WMMA GEMM body: out[M,512] = A[M,512](bf16) * Bt[N][K]^T + bias ----------
// Register-blocked: each wave computes a 32x64 patch (2 row-tiles x 4 col-tiles, 8 accums).
// 8 waves/block cover all 512 columns; blockIdx.x selects the 32-row slab.
template<int OUT_F32>
__device__ __forceinline__ void gemm_body(const __bf16* __restrict__ A,
                                          const __bf16* __restrict__ Bt,   // [N][K] bf16
                                          const float*  __restrict__ bias,
                                          void* __restrict__ outp) {
    const int lane = threadIdx.x & 31;
    const int wave = threadIdx.x >> 5;
    const int row0 = blockIdx.x * 32;
    const int col0 = wave * 64;
    const int l15  = lane & 15;
    const int hi   = (lane >> 4) & 1;

    // A fragment rows: lanes 0-15 carry K = {k0..k0+7, k0+16..k0+23},
    //                  lanes 16-31 carry K = {k0+8..k0+15, k0+24..k0+31}   (ISA 7.12.2)
    const __bf16* arow0 = A + (size_t)(row0 +      l15) * DM + hi * 8;
    const __bf16* arow1 = A + (size_t)(row0 + 16 + l15) * DM + hi * 8;
    // B fragment cols: same K split as A (lanes 0-15 {k0..k0+7, k0+16..k0+23}, lanes 16-31 {k0+8..k0+15, k0+24..k0+31})
    const __bf16* brow0 = Bt + (size_t)(col0 +      l15) * DM + hi * 8;
    const __bf16* brow1 = Bt + (size_t)(col0 + 16 + l15) * DM + hi * 8;
    const __bf16* brow2 = Bt + (size_t)(col0 + 32 + l15) * DM + hi * 8;
    const __bf16* brow3 = Bt + (size_t)(col0 + 48 + l15) * DM + hi * 8;

    v8f acc00 = {}, acc01 = {}, acc02 = {}, acc03 = {};
    v8f acc10 = {}, acc11 = {}, acc12 = {}, acc13 = {};

#pragma unroll
    for (int k0 = 0; k0 < DM; k0 += 32) {
        v16bf a0 = frag(arow0 + k0, arow0 + k0 + 16);
        v16bf a1 = frag(arow1 + k0, arow1 + k0 + 16);
        v16bf b0 = frag(brow0 + k0, brow0 + k0 + 16);
        v16bf b1 = frag(brow1 + k0, brow1 + k0 + 16);
        v16bf b2 = frag(brow2 + k0, brow2 + k0 + 16);
        v16bf b3 = frag(brow3 + k0, brow3 + k0 + 16);
        acc00 = __builtin_amdgcn_wmma_f32_16x16x32_bf16(false, a0, false, b0, (short)0, acc00, false, false);
        acc01 = __builtin_amdgcn_wmma_f32_16x16x32_bf16(false, a0, false, b1, (short)0, acc01, false, false);
        acc02 = __builtin_amdgcn_wmma_f32_16x16x32_bf16(false, a0, false, b2, (short)0, acc02, false, false);
        acc03 = __builtin_amdgcn_wmma_f32_16x16x32_bf16(false, a0, false, b3, (short)0, acc03, false, false);
        acc10 = __builtin_amdgcn_wmma_f32_16x16x32_bf16(false, a1, false, b0, (short)0, acc10, false, false);
        acc11 = __builtin_amdgcn_wmma_f32_16x16x32_bf16(false, a1, false, b1, (short)0, acc11, false, false);
        acc12 = __builtin_amdgcn_wmma_f32_16x16x32_bf16(false, a1, false, b2, (short)0, acc12, false, false);
        acc13 = __builtin_amdgcn_wmma_f32_16x16x32_bf16(false, a1, false, b3, (short)0, acc13, false, false);
    }

    // C/D layout: VGPR r -> row (tileRow + hi*8 + r), col = tileCol + l15
    v8f accs0[4] = {acc00, acc01, acc02, acc03};
    v8f accs1[4] = {acc10, acc11, acc12, acc13};
#pragma unroll
    for (int cb = 0; cb < 4; ++cb) {
        const int n = col0 + cb * 16 + l15;
        const float bn = bias[n];
#pragma unroll
        for (int r = 0; r < 8; ++r) {
            int rowA = row0 + hi * 8 + r;
            int rowB = row0 + 16 + hi * 8 + r;
            float v0 = accs0[cb][r] + bn;
            float v1 = accs1[cb][r] + bn;
            if (OUT_F32) {
                ((float*)outp)[(size_t)rowA * DM + n] = v0;
                ((float*)outp)[(size_t)rowB * DM + n] = v1;
            } else {
                ((__bf16*)outp)[(size_t)rowA * DM + n] = (__bf16)v0;
                ((__bf16*)outp)[(size_t)rowB * DM + n] = (__bf16)v1;
            }
        }
    }
}

// Fused q/k/v projection: blockIdx.y in {0,1,2} selects weight matrix, bias, output slab.
__global__ void qkv_gemm_wmma(const __bf16* __restrict__ A,
                              const __bf16* __restrict__ wt,   // 3 matrices, [N][K] each
                              const float*  __restrict__ bq,
                              const float*  __restrict__ bk,
                              const float*  __restrict__ bv,
                              __bf16* __restrict__ qkvout) {
    const int m = blockIdx.y;
    const __bf16* Bt  = wt + (size_t)m * DM * DM;
    const float* bias = (m == 0) ? bq : (m == 1) ? bk : bv;
    __bf16* outp      = qkvout + (size_t)m * Mm * DM;
    gemm_body<0>(A, Bt, bias, (void*)outp);
}

// Output projection: fp32 result straight to d_out.
__global__ void out_gemm_wmma(const __bf16* __restrict__ A,
                              const __bf16* __restrict__ Bt,
                              const float*  __restrict__ bias,
                              float* __restrict__ outp) {
    gemm_body<1>(A, Bt, bias, (void*)outp);
}

// ---------------- windowed attention (flash-style, one thread per (b,h,q)) ----------------
__global__ void attn_window(const __bf16* __restrict__ qb,
                            const __bf16* __restrict__ kb,
                            const __bf16* __restrict__ vb,
                            __bf16* __restrict__ ctx) {
    int idx = blockIdx.x * blockDim.x + threadIdx.x;    // 65536 = B*H*L
    if (idx >= Bb * Hh * Ll) return;
    int i  = idx & (Ll - 1);
    int bh = idx >> 11;
    int h  = bh & (Hh - 1);
    int b  = bh >> 3;

    size_t qoff = ((size_t)(b * Ll + i) * DM) + h * DK;
    float qv[DK];
#pragma unroll
    for (int t = 0; t < 8; ++t) {
        v8bf x = load8(qb + qoff + t * 8);
#pragma unroll
        for (int e = 0; e < 8; ++e) qv[t * 8 + e] = (float)x[e];
    }

    int jlo = (i - WIN < 0) ? 0 : i - WIN;
    int jhi = (i + WIN > Ll - 1) ? Ll - 1 : i + WIN;

    // warm the cache for the first rows of the window
    {
        size_t koff0 = ((size_t)(b * Ll + jlo) * DM) + h * DK;
        __builtin_prefetch(kb + koff0, 0, 1);   // -> global_prefetch_b8
        __builtin_prefetch(vb + koff0, 0, 1);
    }

    float m = -3.4e38f, sum = 0.0f;
    float cv[DK];
#pragma unroll
    for (int d = 0; d < DK; ++d) cv[d] = 0.0f;

    for (int j = jlo; j <= jhi; ++j) {
        size_t koff = ((size_t)(b * Ll + j) * DM) + h * DK;
        if (j < jhi) {  // prefetch next iteration's K/V rows
            __builtin_prefetch(kb + koff + DM, 0, 1);
            __builtin_prefetch(vb + koff + DM, 0, 1);
        }
        float s = 0.0f;
#pragma unroll
        for (int t = 0; t < 8; ++t) {
            v8bf x = load8(kb + koff + t * 8);
#pragma unroll
            for (int e = 0; e < 8; ++e) s += qv[t * 8 + e] * (float)x[e];
        }
        s *= SCALE;
        s = fminf(fmaxf(s, -CLAMP), CLAMP);

        float mnew = fmaxf(m, s);
        float corr = __expf(m - mnew);
        float p    = __expf(s - mnew);
        sum = sum * corr + p;
        m = mnew;
#pragma unroll
        for (int t = 0; t < 8; ++t) {
            v8bf x = load8(vb + koff + t * 8);
#pragma unroll
            for (int e = 0; e < 8; ++e)
                cv[t * 8 + e] = cv[t * 8 + e] * corr + p * (float)x[e];
        }
    }

    float inv = 1.0f / sum;
#pragma unroll
    for (int t = 0; t < 8; ++t) {
        v8bf y;
#pragma unroll
        for (int e = 0; e < 8; ++e) y[e] = (__bf16)(cv[t * 8 + e] * inv);
        store8(ctx + qoff + t * 8, y);
    }
}

// ---------------- host side ----------------
extern "C" void kernel_launch(void* const* d_in, const int* in_sizes, int n_in,
                              void* d_out, int out_size, void* d_ws, size_t ws_size,
                              hipStream_t stream) {
    const float* Q  = (const float*)d_in[0];
    const float* Wq = (const float*)d_in[1];
    const float* bq = (const float*)d_in[2];
    const float* Wk = (const float*)d_in[3];
    const float* bk = (const float*)d_in[4];
    const float* Wv = (const float*)d_in[5];
    const float* bv = (const float*)d_in[6];
    const float* Wo = (const float*)d_in[7];
    const float* bo = (const float*)d_in[8];
    float* out = (float*)d_out;

    // workspace layout (bf16 elements)
    __bf16* qbf  = (__bf16*)d_ws;              // M*512 activations in bf16
    __bf16* wt   = qbf + (size_t)Mm * DM;      // 4 * 512*512 transposed weights (q,k,v,o)
    __bf16* wtq  = wt;
    __bf16* wtk  = wt + (size_t)DM * DM;
    __bf16* wtv  = wt + (size_t)2 * DM * DM;
    __bf16* wto  = wt + (size_t)3 * DM * DM;
    __bf16* qkv  = wt + (size_t)4 * DM * DM;   // 3 * M*512 (q,k,v)
    __bf16* qout = qkv;
    __bf16* kout = qkv + (size_t)Mm * DM;
    __bf16* vout = qkv + (size_t)2 * Mm * DM;
    __bf16* ctx  = qkv + (size_t)3 * Mm * DM;  // M*512 attention context

    // 1) Q -> bf16
    {
        int n = Mm * DM;
        cvt_f32_bf16<<<(n + 255) / 256, 256, 0, stream>>>(Q, qbf, n);
    }
    // 2) weight transpose+convert
    {
        int blocks = (DM * DM) / 256;
        transpose_cvt<<<blocks, 256, 0, stream>>>(Wq, wtq);
        transpose_cvt<<<blocks, 256, 0, stream>>>(Wk, wtk);
        transpose_cvt<<<blocks, 256, 0, stream>>>(Wv, wtv);
        transpose_cvt<<<blocks, 256, 0, stream>>>(Wo, wto);
    }
    // 3) fused q/k/v projections (WMMA bf16, fp32 accum, bf16 out)
    {
        dim3 grid(Mm / 32, 3);
        qkv_gemm_wmma<<<grid, 256, 0, stream>>>(qbf, wt, bq, bk, bv, qkv);
    }
    // 4) sliding-window attention
    {
        int total = Bb * Hh * Ll;
        attn_window<<<(total + 255) / 256, 256, 0, stream>>>(qout, kout, vout, ctx);
    }
    // 5) output projection -> d_out (fp32)
    {
        dim3 grid(Mm / 32);
        out_gemm_wmma<<<grid, 256, 0, stream>>>(ctx, wto, bo, out);
    }
}
